// SpGAT_50414326120604
// MI455X (gfx1250) — matrix operand order repaired and checked by reading, not compiled, not run on hardware
//
#include <hip/hip_runtime.h>
#include <hip/hip_bf16.h>

// ---- problem constants (match reference) ----
#define NN     50000
#define EE     800000
#define ENN    100000
#define ET_    (EE + ENN)      // 900000 total edges
#define NRELS  500
#define DIM    128             // all feature dims are 128 (NFEAT, RDIM, OUT, 2*NHID)
#define ALPHA_ 0.2f

typedef __attribute__((ext_vector_type(16))) __bf16 v16bf;
typedef __attribute__((ext_vector_type(8)))  float  v8f;

__device__ __forceinline__ float elu1(float v) {
    return v > 0.f ? v : (__expf(v) - 1.f);
}

// ---------------------------------------------------------------------------
// zero-fill (grid-stride)
// ---------------------------------------------------------------------------
__global__ void zero_f32(float* __restrict__ p, size_t n) {
    size_t i = (size_t)blockIdx.x * blockDim.x + threadIdx.x;
    size_t stride = (size_t)gridDim.x * blockDim.x;
    for (; i < n; i += stride) p[i] = 0.f;
}

// ---------------------------------------------------------------------------
// C[M,128] = A[M,128] (f32, row-major) x W^T, with W element (n,k) fetched as
//   W[wofs + n*sN + k*SK]   (SK is a compile-time constant)
// SK==1  -> weight stored [dout, din] row-major: vectorized float4 loads
// SK==ld -> weight stored [din, dout]: scalar loads (only used for tiny GEMMs)
// One 16x16 output tile per wave; 8 waves/block share a bf16 A-tile in LDS.
// K = 128 = 4 x v_wmma_f32_16x16x32_bf16.
// ---------------------------------------------------------------------------
template <int SK>
__global__ __launch_bounds__(256) void gemm_x128(
    const float* __restrict__ A, int M,
    const float* __restrict__ W, int sN, int wofs,
    float* __restrict__ C)
{
    __shared__ __bf16 As[16][136];   // 16 rows x 128 bf16, +8 pad vs bank conflicts

    const int tid   = threadIdx.x;
    const int mtile = blockIdx.x;

    // cooperative load of the 16x128 A tile (each thread: 8 consecutive f32)
    {
        int elem = tid * 8;
        int r = elem >> 7;
        int c = elem & 127;
        int grow = mtile * 16 + r;
        if (grow >= M) grow = M - 1;              // clamp: padded rows never stored
        const float* src = A + (size_t)grow * DIM + c;
        float4 v0 = *(const float4*)(src);
        float4 v1 = *(const float4*)(src + 4);
        __bf16* dst = &As[r][c];
        dst[0] = (__bf16)v0.x; dst[1] = (__bf16)v0.y;
        dst[2] = (__bf16)v0.z; dst[3] = (__bf16)v0.w;
        dst[4] = (__bf16)v1.x; dst[5] = (__bf16)v1.y;
        dst[6] = (__bf16)v1.z; dst[7] = (__bf16)v1.w;
    }
    __syncthreads();

    const int wave = tid >> 5;           // 8 waves -> 8 N-tiles of 16 (dout = 128)
    const int lane = tid & 31;
    const int n0   = wave * 16;
    const int m    = lane & 15;
    const int hi   = lane >> 4;          // half-wave select
    const int koff = hi * 16;            // A: lane>=16 holds K 16..31
    const int kbB  = hi * 16;            // B: lane>=16 holds K 16..31
    const int n    = n0 + m;

    const float* Wp = W + wofs + (size_t)n * sN;

    // ---- preload all four B fragments (K blocks of 32) ----
    v16bf bfrag[4];
#pragma unroll
    for (int f = 0; f < 4; ++f) {
        const float* wk = Wp + (size_t)(f * 32 + kbB) * SK;
        if (SK == 1) {
            // 16 contiguous f32 -> 4x float4, cvt to bf16
            float4 w0 = *(const float4*)(wk + 0);
            float4 w1 = *(const float4*)(wk + 4);
            float4 w2 = *(const float4*)(wk + 8);
            float4 w3 = *(const float4*)(wk + 12);
            bfrag[f][0]  = (__bf16)w0.x; bfrag[f][1]  = (__bf16)w0.y;
            bfrag[f][2]  = (__bf16)w0.z; bfrag[f][3]  = (__bf16)w0.w;
            bfrag[f][4]  = (__bf16)w1.x; bfrag[f][5]  = (__bf16)w1.y;
            bfrag[f][6]  = (__bf16)w1.z; bfrag[f][7]  = (__bf16)w1.w;
            bfrag[f][8]  = (__bf16)w2.x; bfrag[f][9]  = (__bf16)w2.y;
            bfrag[f][10] = (__bf16)w2.z; bfrag[f][11] = (__bf16)w2.w;
            bfrag[f][12] = (__bf16)w3.x; bfrag[f][13] = (__bf16)w3.y;
            bfrag[f][14] = (__bf16)w3.z; bfrag[f][15] = (__bf16)w3.w;
        } else {
#pragma unroll
            for (int e = 0; e < 16; ++e)
                bfrag[f][e] = (__bf16)wk[e * SK];
        }
    }

    v8f acc = {0.f, 0.f, 0.f, 0.f, 0.f, 0.f, 0.f, 0.f};

#pragma unroll
    for (int f = 0; f < 4; ++f) {        // K blocks of 32
        v16bf a;
        const __bf16* arow = &As[m][f * 32];
#pragma unroll
        for (int e = 0; e < 8; ++e) {
            a[e]     = arow[koff + e];            // K = f*32 + koff + e
            a[8 + e] = arow[8 + koff + e];        // K = f*32 + 8 + koff + e
        }
        acc = __builtin_amdgcn_wmma_f32_16x16x32_bf16(
                  false, a, false, bfrag[f], (short)0, acc, false, false);
    }

    // store D: VGPR r -> row (r + hi*8), col n
    float* crow = C + (size_t)(mtile * 16 + hi * 8) * DIM + n;
    if ((mtile + 1) * 16 <= M) {
        // wave-uniform fast path: whole tile in bounds, no per-row guards
#pragma unroll
        for (int r = 0; r < 8; ++r)
            crow[(size_t)r * DIM] = acc[r];
    } else {
#pragma unroll
        for (int r = 0; r < 8; ++r) {
            int gr = mtile * 16 + r + hi * 8;
            if (gr < M) C[(size_t)gr * DIM + n] = acc[r];
        }
    }
}

// ---------------------------------------------------------------------------
// Layer-1 edge pass: one wave per edge; lanes 0-15 -> head 0, 16-31 -> head 1
// (4 f32 per lane). Per-head 16-lane butterfly dot, exp(-leaky_relu),
// atomic scatter into rowsum[ s*2+h ] and hnum[ s*128 + h*64 + j ].
// ---------------------------------------------------------------------------
__global__ __launch_bounds__(256) void edge_pass1(
    const int* __restrict__ el,   const int* __restrict__ eln,
    const int* __restrict__ etn,
    const float* __restrict__ pm_src, const float* __restrict__ pm_dst,
    const float* __restrict__ relproj, const float* __restrict__ relrel,
    const float* __restrict__ a2,
    float* __restrict__ hnum, float* __restrict__ rowsum)
{
    int w = (int)((blockIdx.x * (unsigned)blockDim.x + threadIdx.x) >> 5);
    if (w >= ET_) return;
    int lane = threadIdx.x & 31;

    int s, d;
    float4 rel;
    if (w < EE) {
        s = el[w];  d = el[EE + w];
        rel = *(const float4*)(relproj + (size_t)w * DIM + lane * 4);
    } else {
        int u = w - EE;
        s = eln[u]; d = eln[ENN + u];
        int t0 = etn[2 * u], t1 = etn[2 * u + 1];
        float4 r0 = *(const float4*)(relrel + (size_t)t0 * DIM + lane * 4);
        float4 r1 = *(const float4*)(relrel + (size_t)t1 * DIM + lane * 4);
        rel = make_float4(r0.x + r1.x, r0.y + r1.y, r0.z + r1.z, r0.w + r1.w);
    }

    float4 ms = *(const float4*)(pm_src + (size_t)s * DIM + lane * 4);
    float4 md = *(const float4*)(pm_dst + (size_t)d * DIM + lane * 4);
    float4 m  = make_float4(ms.x + md.x + rel.x, ms.y + md.y + rel.y,
                            ms.z + md.z + rel.z, ms.w + md.w + rel.w);
    float4 av = *(const float4*)(a2 + lane * 4);

    float p = m.x * av.x + m.y * av.y + m.z * av.z + m.w * av.w;
    // reduce within each 16-lane (per-head) group
    p += __shfl_xor(p, 1, 32);
    p += __shfl_xor(p, 2, 32);
    p += __shfl_xor(p, 4, 32);
    p += __shfl_xor(p, 8, 32);

    float lr = p > 0.f ? p : ALPHA_ * p;
    float ee = __expf(-lr);

    if ((lane & 15) == 0)
        atomicAdd(rowsum + (size_t)s * 2 + (lane >> 4), ee);

    float* hp = hnum + (size_t)s * DIM + lane * 4;
    atomicAdd(hp + 0, ee * m.x);
    atomicAdd(hp + 1, ee * m.y);
    atomicAdd(hp + 2, ee * m.z);
    atomicAdd(hp + 3, ee * m.w);
}

// ---------------------------------------------------------------------------
// Layer-2 edge pass: one wave per edge, full 128-wide single head.
// Also accumulates raw h_num (-> out2) and raw h_sum (-> out3) plus cnt.
// ---------------------------------------------------------------------------
__global__ __launch_bounds__(256) void edge_pass2(
    const int* __restrict__ el,  const int* __restrict__ eln,
    const int* __restrict__ et,  const int* __restrict__ etn,
    const float* __restrict__ pm_src, const float* __restrict__ pm_dst,
    const float* __restrict__ relproj2, const float* __restrict__ a2,
    float* __restrict__ hnum, float* __restrict__ hsum,
    float* __restrict__ rowsum, float* __restrict__ cnt)
{
    int w = (int)((blockIdx.x * (unsigned)blockDim.x + threadIdx.x) >> 5);
    if (w >= ET_) return;
    int lane = threadIdx.x & 31;

    int s, d;
    float4 rel;
    if (w < EE) {
        s = el[w];  d = el[EE + w];
        int r = et[w];
        rel = *(const float4*)(relproj2 + (size_t)r * DIM + lane * 4);
    } else {
        int u = w - EE;
        s = eln[u]; d = eln[ENN + u];
        int t0 = etn[2 * u], t1 = etn[2 * u + 1];
        float4 r0 = *(const float4*)(relproj2 + (size_t)t0 * DIM + lane * 4);
        float4 r1 = *(const float4*)(relproj2 + (size_t)t1 * DIM + lane * 4);
        rel = make_float4(r0.x + r1.x, r0.y + r1.y, r0.z + r1.z, r0.w + r1.w);
    }

    float4 ms = *(const float4*)(pm_src + (size_t)s * DIM + lane * 4);
    float4 md = *(const float4*)(pm_dst + (size_t)d * DIM + lane * 4);
    float4 m  = make_float4(ms.x + md.x + rel.x, ms.y + md.y + rel.y,
                            ms.z + md.z + rel.z, ms.w + md.w + rel.w);
    float4 av = *(const float4*)(a2 + lane * 4);

    float p = m.x * av.x + m.y * av.y + m.z * av.z + m.w * av.w;
    p += __shfl_xor(p, 1, 32);
    p += __shfl_xor(p, 2, 32);
    p += __shfl_xor(p, 4, 32);
    p += __shfl_xor(p, 8, 32);
    p += __shfl_xor(p, 16, 32);

    float lr = p > 0.f ? p : ALPHA_ * p;
    float ee = __expf(-lr);

    if (lane == 0) {
        atomicAdd(rowsum + s, ee);
        atomicAdd(cnt + s, 1.f);
    }

    float* hp = hnum + (size_t)s * DIM + lane * 4;
    atomicAdd(hp + 0, ee * m.x);
    atomicAdd(hp + 1, ee * m.y);
    atomicAdd(hp + 2, ee * m.z);
    atomicAdd(hp + 3, ee * m.w);
    float* up = hsum + (size_t)s * DIM + lane * 4;
    atomicAdd(up + 0, m.x);
    atomicAdd(up + 1, m.y);
    atomicAdd(up + 2, m.z);
    atomicAdd(up + 3, m.w);
}

// ---------------------------------------------------------------------------
// finalize layer 1 in place: x2 = elu(h_num1 / rowsum1)
// ---------------------------------------------------------------------------
__global__ void fin1(float* __restrict__ hx, const float* __restrict__ rowsum)
{
    size_t i = (size_t)blockIdx.x * blockDim.x + threadIdx.x;
    if (i >= (size_t)NN * DIM) return;
    int node = (int)(i >> 7);
    int j = (int)(i & 127);
    float rs = rowsum[(size_t)node * 2 + (j >> 6)];
    if (rs == 0.f) rs = 1e-12f;
    float v = hx[i] / rs;
    hx[i] = elu1(v);
}

// ---------------------------------------------------------------------------
// finalize layer 2:
//   out0 = elu(h_num2/rowsum2); out2 = elu(h_num2) in place; out3 = elu(h_sum2/max(cnt,1))
// ---------------------------------------------------------------------------
__global__ void fin2(float* __restrict__ out0, float* __restrict__ out2,
                     float* __restrict__ out3,
                     const float* __restrict__ rowsum, const float* __restrict__ cnt)
{
    size_t i = (size_t)blockIdx.x * blockDim.x + threadIdx.x;
    if (i >= (size_t)NN * DIM) return;
    int node = (int)(i >> 7);
    float rs = rowsum[node];
    if (rs == 0.f) rs = 1e-12f;
    float c = cnt[node];
    if (c < 1.f) c = 1.f;
    float hn = out2[i];
    float hs = out3[i];
    out0[i] = elu1(hn / rs);
    out2[i] = elu1(hn);
    out3[i] = elu1(hs / c);
}

// ---------------------------------------------------------------------------
extern "C" void kernel_launch(void* const* d_in, const int* in_sizes, int n_in,
                              void* d_out, int out_size, void* d_ws, size_t ws_size,
                              hipStream_t stream)
{
    const float* x    = (const float*)d_in[0];   // [N,128]
    const float* rel  = (const float*)d_in[1];   // [500,128]
    const float* eemb = (const float*)d_in[2];   // [E,128]
    const float* ah   = (const float*)d_in[3];   // [2,64,384] -> row n, ld 384
    const float* a2h  = (const float*)d_in[4];   // [2,64] -> flat 128
    const float* Wm   = (const float*)d_in[5];   // [128,128] (din,dout)
    const float* aout = (const float*)d_in[6];   // [128,384]
    const float* a2o  = (const float*)d_in[7];   // [128]
    const int*   el   = (const int*)d_in[8];     // [2,E]
    const int*   et   = (const int*)d_in[9];     // [E]
    const int*   eln  = (const int*)d_in[10];    // [2,EN]
    const int*   etn  = (const int*)d_in[11];    // [EN,2]

    float* out  = (float*)d_out;
    float* out0 = out;                                   // elu(h)      [N,128]
    float* out1 = out0 + (size_t)NN * DIM;               // out_rel     [500,128]
    float* out2 = out1 + (size_t)NRELS * DIM;            // elu(h_num2) [N,128]
    float* out3 = out2 + (size_t)NN * DIM;               // elu(h_unif) [N,128]

    float* w = (float*)d_ws;
    float* pm1s     = w;  w += (size_t)NN * DIM;      // layer1 src proj (both heads)
    float* pm1d     = w;  w += (size_t)NN * DIM;      // layer1 dst proj
    float* relrel1  = w;  w += (size_t)NRELS * DIM;   // rel-table proj, layer1 (nhop)
    float* relproj2 = w;  w += (size_t)NRELS * DIM;   // rel-table proj, layer2
    float* hnum1    = w;  w += (size_t)NN * DIM;      // accum -> becomes x2 in place
    float* rowsum1  = w;  w += (size_t)NN * 2;
    float* rowsum2  = w;  w += (size_t)NN;
    float* cnt2     = w;  w += (size_t)NN;
    float* relproj1 = w;  w += (size_t)EE * DIM;      // per-edge rel proj, layer1
    float* pm2s     = pm1s;                            // reuse after fin1
    float* pm2d     = pm1d;

    // zero the atomic accumulators (hnum1|rowsum1|rowsum2|cnt2 contiguous) and
    // the d_out regions used as layer-2 accumulators (out2|out3 contiguous)
    zero_f32<<<2048, 256, 0, stream>>>(hnum1, (size_t)NN * (DIM + 4));
    zero_f32<<<2048, 256, 0, stream>>>(out2, (size_t)NN * DIM * 2);

    auto mtiles = [](int M) { return (M + 15) / 16; };

    // layer-1 dense projections (WMMA, contiguous-K weights)
    gemm_x128<1><<<mtiles(NN),    256, 0, stream>>>(x,    NN,    ah, 384,   0, pm1s);
    gemm_x128<1><<<mtiles(NN),    256, 0, stream>>>(x,    NN,    ah, 384, 128, pm1d);
    gemm_x128<1><<<mtiles(NRELS), 256, 0, stream>>>(rel,  NRELS, ah, 384, 256, relrel1);
    gemm_x128<1><<<mtiles(EE),    256, 0, stream>>>(eemb, EE,    ah, 384, 256, relproj1);

    // out_rel = rel @ W  (weight stored [din,dout]: sN=1, SK=128) -- tiny GEMM
    gemm_x128<128><<<mtiles(NRELS), 256, 0, stream>>>(rel, NRELS, Wm, 1, 0, out1);
    // layer-2 relation projection of the 500-row table (linearity of a_out_rel)
    gemm_x128<1><<<mtiles(NRELS), 256, 0, stream>>>(out1, NRELS, aout, 384, 256, relproj2);

    // layer-1 sparse attention (one wave per edge)
    edge_pass1<<<(ET_ * 32) / 256, 256, 0, stream>>>(
        el, eln, etn, pm1s, pm1d, relproj1, relrel1, a2h, hnum1, rowsum1);
    fin1<<<(int)(((size_t)NN * DIM + 255) / 256), 256, 0, stream>>>(hnum1, rowsum1);

    // layer-2 dense projections (x2 lives in hnum1)
    gemm_x128<1><<<mtiles(NN), 256, 0, stream>>>(hnum1, NN, aout, 384,   0, pm2s);
    gemm_x128<1><<<mtiles(NN), 256, 0, stream>>>(hnum1, NN, aout, 384, 128, pm2d);

    // layer-2 sparse attention
    edge_pass2<<<(ET_ * 32) / 256, 256, 0, stream>>>(
        el, eln, et, etn, pm2s, pm2d, relproj2, a2o, out2, out3, rowsum2, cnt2);
    fin2<<<(int)(((size_t)NN * DIM + 255) / 256), 256, 0, stream>>>(
        out0, out2, out3, rowsum2, cnt2);
}
